// ECGSegMCULBit_19310172962891
// MI455X (gfx1250) — matrix operand order repaired and checked by reading, not compiled, not run on hardware
//
#include <hip/hip_runtime.h>
#include <hip/hip_bf16.h>

typedef __attribute__((ext_vector_type(16))) _Float16 v16h;
typedef __attribute__((ext_vector_type(8)))  _Float16 v8h;
typedef __attribute__((ext_vector_type(4)))  _Float16 v4h;
typedef __attribute__((ext_vector_type(8)))  float    v8f;

// ---------------- quantization helpers (exact per reference) ----------------
__device__ __forceinline__ float lbitf(float x)  { return rintf(x * 64.0f) * (1.0f / 64.0f); }
__device__ __forceinline__ float qtanhf(float x) { return rintf(tanhf(x) * 64.0f) * (1.0f / 64.0f); }

// Output modes for conv kernels:
//   0 = R32 : fp32 [Cout][Lout], lbit values        (consumer: pool/softmax/finalize)
//   1 = T4  : f16  [Lout][4],   qtanh(lbit) values  (consumer: Cin=4 k=3 conv)
//   2 = T32P: f16  [Lout][32],  qtanh(lbit), cols Cout..31 zeroed (consumer: Cin=8 k=1 conv)

// ---- weight prep: fp32 -> f16 A-fragments in per-lane contiguous order -----
// dst layout: [wt][chunk][lane 0..31][hh 0..15]  (one 32B v16h per lane/chunk)
__global__ void ecg_prep_wfrag(const float* __restrict__ w, _Float16* __restrict__ dst,
                               int Cout, int Cin, int Kw, int WT, int KC, int bin) {
  int idx = blockIdx.x * blockDim.x + threadIdx.x;
  int total = WT * KC * 512;
  if (idx >= total) return;
  int hh   = idx & 15;
  int lane = (idx >> 4) & 31;
  int c    = (idx >> 9) % KC;
  int wt   = (idx >> 9) / KC;
  int m = lane & 15, laneHi = (lane >> 4) & 1;
  int j = hh >> 1, p = hh & 1;
  int ka = (laneHi << 4) + hh;
  int kk = c * 32 + ka;
  int co = wt * 16 + m;
  float v = 0.0f;
  if (co < Cout && kk < Cin * Kw) {
    int ci = kk / Kw, k = kk - ci * Kw;
    float raw = w[((size_t)co * Cin + ci) * Kw + k];
    v = bin ? (raw >= 0.0f ? 1.0f : -1.0f) : qtanhf(raw);
  }
  dst[idx] = (_Float16)v;
}

// ---- front: pad(236,100)+/8+lbit+avgpool(4,4)+lbit+qtanh -> f16 [L=1334][32]
__global__ void ecg_front(const float* __restrict__ x, _Float16* __restrict__ outT, int N) {
  const int Lout = 1334, C = 12, Lsrc = 5000, PADL = 236;
  size_t total = (size_t)N * Lout * 32;
  size_t gid = (size_t)blockIdx.x * blockDim.x + threadIdx.x;
  if (gid >= total) return;
  int c = (int)(gid & 31);
  size_t rest = gid >> 5;
  int i = (int)(rest % Lout), n = (int)(rest / Lout);
  float v = 0.0f;
  if (c < C) {
    const float* row = x + ((size_t)n * C + c) * Lsrc;
    float s = 0.0f;
    #pragma unroll
    for (int t = 0; t < 4; ++t) {
      int p = 4 * i + t - PADL;
      float u = (p >= 0 && p < Lsrc) ? row[p] * 0.125f : 0.0f;
      s += lbitf(u);
    }
    v = qtanhf(lbitf(s * 0.25f));
  }
  outT[gid] = (_Float16)v;
}

// ---- shared D-writeback for all conv kernels -------------------------------
__device__ __forceinline__ void conv_store(float* accf, void* outv, int n, int wv, int laneHi,
                                           int pos, int Cout, int Lout, int omode) {
  if (pos >= Lout) return;
  if (omode == 0) {
    float* out32 = (float*)outv;
    #pragma unroll
    for (int r = 0; r < 8; ++r) {
      int co = wv * 16 + (laneHi << 3) + r;
      if (co < Cout) out32[((size_t)n * Cout + co) * Lout + pos] = lbitf(accf[r]);
    }
  } else if (omode == 1) {          // T4 (Cout==4, WT==1)
    if (laneHi == 0) {
      _Float16* row = (_Float16*)outv + ((size_t)n * Lout + pos) * 4;
      v4h q;
      #pragma unroll
      for (int r = 0; r < 4; ++r) q[r] = (_Float16)qtanhf(lbitf(accf[r]));
      *(v4h*)row = q;
    }
  } else {                          // T32P (Cout<=8, WT==1, pitch 32 zero-padded)
    _Float16* row = (_Float16*)outv + ((size_t)n * Lout + pos) * 32;
    v8h q;
    #pragma unroll
    for (int r = 0; r < 8; ++r) {
      int co = (laneHi << 3) + r;
      q[r] = (co < Cout) ? (_Float16)qtanhf(lbitf(accf[r])) : (_Float16)0.0f;
    }
    *(v8h*)(row + (laneHi << 3)) = q;
    v8h z = {};
    *(v8h*)(row + 16 + (laneHi << 3)) = z;
  }
}

// ---- direct conv (Kw=1): B fragments straight from global qt16T input ------
// inT: f16 [Lin][CP] (CP = K-pad of this layer, rows already qtanh'd, padded w/ 0)
template <int CP, int WT>
__global__ void ecg_qconv_direct(const _Float16* __restrict__ inT, const _Float16* __restrict__ wfrag,
                                 void* __restrict__ outv, int Cout, int Lin, int Lout, int omode) {
  constexpr int KC = CP / 32;
  const int n  = blockIdx.y;
  const int l0 = blockIdx.x * 16;
  const int tid = threadIdx.x;
  const int wv = tid >> 5, lane = tid & 31;
  const int laneHi = (lane >> 4) & 1, m = lane & 15;

  // row of qt16 K-values for this lane's output position (over-read at tail is masked)
  const _Float16* bbase = inT + ((size_t)n * Lin + l0 + m) * CP;
  __builtin_prefetch(bbase + 16 * CP, 0, 1);   // warm L2 for the adjacent tile
  const _Float16* wbase = wfrag + (size_t)wv * (KC * 512);

  union { v8f v; float f[8]; } acc;
  #pragma unroll
  for (int r = 0; r < 8; ++r) acc.f[r] = 0.0f;

  #pragma unroll
  for (int c = 0; c < KC; ++c) {
    v16h a = *(const v16h*)(wbase + ((size_t)(c * 32 + lane) << 4));
    union { v16h v; v8h h8[2]; } b;
    b.h8[0] = *(const v8h*)(bbase + c * 32 + laneHi * 16);
    b.h8[1] = *(const v8h*)(bbase + c * 32 + laneHi * 16 + 8);
    acc.v = __builtin_amdgcn_wmma_f32_16x16x32_f16(false, a, false, b.v,
                                                   (short)0, acc.v, false, false);
  }
  conv_store(acc.f, outv, n, wv, laneHi, l0 + m, Cout, Lout, omode);
}

// ---- k=3 conv (Cin=4): LDS im2col from qt16T pitch-4 input (pure f16 copy) -
template <int WT>
__global__ void ecg_qconv_k3(const _Float16* __restrict__ inT, const _Float16* __restrict__ wfrag,
                             void* __restrict__ outv, int Cout, int Lin, int Lout, int omode) {
  constexpr int CINKW = 12, KPAD = 32, SBS = 40;
  __shared__ __align__(32) _Float16 s_b[16][SBS];
  const int n  = blockIdx.y;
  const int l0 = blockIdx.x * 16;
  const int tid = threadIdx.x, nthr = blockDim.x;

  for (int idx = tid; idx < 16 * KPAD; idx += nthr) {
    int pos = idx & 15;
    int kk  = idx >> 4;
    _Float16 h = (_Float16)0.0f;
    if (kk < CINKW) {
      int ci = kk / 3, k = kk - ci * 3;
      int l = l0 + pos + k;
      if (l < Lin) h = inT[((size_t)n * Lin + l) * 4 + ci];
    }
    s_b[pos][kk] = h;
  }
  __syncthreads();

  const int wv = tid >> 5, lane = tid & 31;
  const int laneHi = (lane >> 4) & 1, m = lane & 15;
  union { v8f v; float f[8]; } acc;
  #pragma unroll
  for (int r = 0; r < 8; ++r) acc.f[r] = 0.0f;

  v16h a = *(const v16h*)(wfrag + (size_t)wv * 512 + ((size_t)lane << 4));
  union { v16h v; v8h h8[2]; } b;
  b.h8[0] = *(const v8h*)&s_b[m][laneHi * 16];
  b.h8[1] = *(const v8h*)&s_b[m][laneHi * 16 + 8];
  acc.v = __builtin_amdgcn_wmma_f32_16x16x32_f16(false, a, false, b.v,
                                                 (short)0, acc.v, false, false);
  conv_store(acc.f, outv, n, wv, laneHi, l0 + m, Cout, Lout, omode);
}

// ---- upsample conv (Cin=64, k=1, Cout=8): fp32 raw in, fused interp+qtanh --
__global__ void ecg_qconv_ups(const float* __restrict__ in, const _Float16* __restrict__ wfrag,
                              _Float16* __restrict__ outT, int Cout, int Lout,
                              int ups_scale, int ups_n) {
  constexpr int CIN = 64, KC = 2, SBS = 72;
  __shared__ __align__(32) _Float16 s_b[16][SBS];
  const int n  = blockIdx.y;
  const int l0 = blockIdx.x * 16;
  const int tid = threadIdx.x;

  for (int idx = tid; idx < 16 * CIN; idx += 32) {
    int pos = idx & 15;
    int ci  = idx >> 4;
    float v = 0.0f;
    int l = l0 + pos;
    if (l < Lout) {
      int j = l + (ups_scale >> 1);
      float src = fmaxf(((float)j + 0.5f) / (float)ups_scale - 0.5f, 0.0f);
      int i0 = (int)floorf(src);
      int i1 = min(i0 + 1, ups_n - 1);
      float wgt = src - (float)i0;
      const float* row = in + ((size_t)n * CIN + ci) * ups_n;
      float a = lbitf(row[i0]), b = lbitf(row[i1]);
      v = qtanhf(lbitf(a * (1.0f - wgt) + b * wgt));
    }
    s_b[pos][ci] = (_Float16)v;
  }
  __syncthreads();

  const int lane = tid & 31;
  const int laneHi = (lane >> 4) & 1, m = lane & 15;
  union { v8f v; float f[8]; } acc;
  #pragma unroll
  for (int r = 0; r < 8; ++r) acc.f[r] = 0.0f;

  #pragma unroll
  for (int c = 0; c < KC; ++c) {
    v16h a = *(const v16h*)(wfrag + ((size_t)(c * 32 + lane) << 4));
    union { v16h v; v8h h8[2]; } b;
    b.h8[0] = *(const v8h*)&s_b[m][c * 32 + laneHi * 16];
    b.h8[1] = *(const v8h*)&s_b[m][c * 32 + laneHi * 16 + 8];
    acc.v = __builtin_amdgcn_wmma_f32_16x16x32_f16(false, a, false, b.v,
                                                   (short)0, acc.v, false, false);
  }
  conv_store(acc.f, (void*)outT, n, 0, laneHi, l0 + m, Cout, Lout, /*T32P*/2);
}

// ---------------- avgpool: lbit(in) -> window mean -> lbit (fp32 [C][L]) ----
__global__ void ecg_avgpool(const float* __restrict__ in, float* __restrict__ out,
                            int N, int C, int Lin, int Lout, int k, int s) {
  size_t total = (size_t)N * C * Lout;
  size_t gid = (size_t)blockIdx.x * blockDim.x + threadIdx.x;
  if (gid >= total) return;
  int i = (int)(gid % Lout);
  size_t rc = gid / Lout;
  const float* row = in + rc * Lin;
  float ssum = 0.0f;
  for (int t = 0; t < k; ++t) ssum += lbitf(row[i * s + t]);
  out[gid] = lbitf(ssum / (float)k);
}

// ---- channel poly-softmax (C=64); omode 0: fp32 [64][L]; 1: f16 [L][64] ----
__global__ void ecg_poly_softmax(const float* __restrict__ in, void* __restrict__ outv,
                                 int L, int omode) {
  const int C = 64;
  int n = blockIdx.y;
  int pos = blockIdx.x * blockDim.x + threadIdx.x;
  if (pos >= L) return;
  const float* p = in + (size_t)n * C * L + pos;
  float mx = -3.0e38f;
  for (int c = 0; c < C; ++c) mx = fmaxf(mx, lbitf(p[(size_t)c * L]));
  float sum = 0.0f;
  for (int c = 0; c < C; ++c) {
    float d = mx - lbitf(p[(size_t)c * L]);
    float d2 = d * d * 0.5f, d3 = d2 * d * (1.0f / 3.0f), d4 = d3 * d * 0.25f;
    sum += lbitf(1.0f / (1.0f + d + d2 + d3 + d4));
  }
  float inv = 1.0f / sum;
  for (int c = 0; c < C; ++c) {
    float d = mx - lbitf(p[(size_t)c * L]);
    float d2 = d * d * 0.5f, d3 = d2 * d * (1.0f / 3.0f), d4 = d3 * d * 0.25f;
    float o = lbitf(lbitf(1.0f / (1.0f + d + d2 + d3 + d4)) * inv);
    if (omode == 0) ((float*)outv)[(size_t)n * C * L + (size_t)c * L + pos] = o;
    else ((_Float16*)outv)[((size_t)n * L + pos) * 64 + c] = (_Float16)qtanhf(o);
  }
}

// ---------------- finalize: fp32 [N,4,5016] --crop 16--> [N,5000,4] ---------
__global__ void ecg_finalize(const float* __restrict__ in, float* __restrict__ out, int N) {
  const int L = 5016, P = 5000, C = 4;
  size_t total = (size_t)N * P;
  size_t gid = (size_t)blockIdx.x * blockDim.x + threadIdx.x;
  if (gid >= total) return;
  int p = (int)(gid % P), n = (int)(gid / P);
  #pragma unroll
  for (int c = 0; c < C; ++c)
    out[gid * C + c] = in[((size_t)n * C + c) * L + 16 + p];
}

// ============================ host side =====================================
struct LayerCfg { int Cout, Cin, Kw, bin; };

extern "C" void kernel_launch(void* const* d_in, const int* in_sizes, int n_in,
                              void* d_out, int out_size, void* d_ws, size_t ws_size,
                              hipStream_t stream) {
  (void)n_in; (void)out_size; (void)ws_size;
  const float* x = (const float*)d_in[0];
  const int N = in_sizes[0] / (12 * 5000);

  static const LayerCfg LC[14] = {
    {4,12,1,0},{4,4,3,0},{64,4,3,1},{4,64,1,1},{4,4,3,0},{64,4,3,1},
    {4,64,1,1},{4,4,3,0},{64,4,3,1},{8,64,1,1},{8,8,1,0},{64,8,1,1},
    {8,64,1,1},{4,8,1,0}
  };
  int woff[14], kc[14], wt[14];
  int acc = 0;
  for (int i = 0; i < 14; ++i) {
    kc[i] = (((LC[i].Cin * LC[i].Kw) + 31) / 32);
    wt[i] = (LC[i].Cout + 15) / 16;
    woff[i] = acc;
    acc += wt[i] * kc[i] * 512;
  }

  char* ws = (char*)d_ws;
  _Float16* wq = (_Float16*)ws;              // 15360 halves < 32 KB
  float* A1 = (float*)(ws + 32768);
  size_t arena = (size_t)N * 64 * 1330;      // largest intermediate (fp32 elems)
  float* A2 = A1 + arena;
  _Float16* H1 = (_Float16*)A1;              // f16 views of the same arenas
  _Float16* H2 = (_Float16*)A2;

  for (int i = 0; i < 14; ++i) {
    int total = wt[i] * kc[i] * 512;
    ecg_prep_wfrag<<<(total + 255) / 256, 256, 0, stream>>>(
        (const float*)d_in[1 + i], wq + woff[i],
        LC[i].Cout, LC[i].Cin, LC[i].Kw, wt[i], kc[i], LC[i].bin);
  }

#define GRID(Lout, WTN) dim3((unsigned)(((Lout) + 15) / 16), (unsigned)N), dim3(32 * (WTN)), 0, stream
  auto pool = [&](const float* in, float* out, int Lin, int Lout, int k, int s) {
    size_t total = (size_t)N * 64 * Lout;
    ecg_avgpool<<<(unsigned)((total + 255) / 256), 256, 0, stream>>>(in, out, N, 64, Lin, Lout, k, s);
  };
  auto softmax = [&](const float* in, void* out, int L, int omode) {
    ecg_poly_softmax<<<dim3((unsigned)((L + 255) / 256), (unsigned)N), 256, 0, stream>>>(in, out, L, omode);
  };

  // front -> H1: f16 [1334][32] (12ch + zero pad), qtanh domain
  {
    size_t total = (size_t)N * 1334 * 32;
    ecg_front<<<(unsigned)((total + 255) / 256), 256, 0, stream>>>(x, H1, N);
  }
  // ---- down path ----
  ecg_qconv_direct<32,1><<<GRID(1334,1)>>>(H1, wq+woff[0], H2, 4, 1334, 1334, 1);  // dw1 -> T4
  ecg_qconv_k3<1>    <<<GRID(1332,1)>>>(H2, wq+woff[1], H1, 4, 1334, 1332, 1);     // dw2 -> T4
  ecg_qconv_k3<4>    <<<GRID(1330,4)>>>(H1, wq+woff[2], A2, 64, 1332, 1330, 0);    // dw3 -> R32
  pool(A2, A1, 1330, 266, 5, 5);
  softmax(A1, H2, 266, 1);                                                          // -> f16 [266][64]
  ecg_qconv_direct<64,1><<<GRID(266,1)>>>(H2, wq+woff[3], H1, 4, 266, 266, 1);     // dw4 -> T4
  ecg_qconv_k3<1>    <<<GRID(264,1)>>>(H1, wq+woff[4], H2, 4, 266, 264, 1);        // dw5 -> T4
  ecg_qconv_k3<4>    <<<GRID(262,4)>>>(H2, wq+woff[5], A1, 64, 264, 262, 0);       // dw6 -> R32
  pool(A1, A2, 262, 260, 3, 1);
  softmax(A2, H1, 260, 1);
  ecg_qconv_direct<64,1><<<GRID(260,1)>>>(H1, wq+woff[6], H2, 4, 260, 260, 1);     // dw7 -> T4
  ecg_qconv_k3<1>    <<<GRID(258,1)>>>(H2, wq+woff[7], H1, 4, 260, 258, 1);        // dw8 -> T4
  ecg_qconv_k3<4>    <<<GRID(256,4)>>>(H1, wq+woff[8], A2, 64, 258, 256, 0);       // dw9 -> R32
  pool(A2, A1, 256, 252, 5, 1);
  softmax(A1, A2, 252, 0);                                                          // fp32 raw for ups
  // ---- up path ----
  ecg_qconv_ups<<<GRID(1255,1)>>>(A2, wq+woff[9],  H1, 8, 1255, 5, 252);           // uw1 -> T32P
  ecg_qconv_direct<32,1><<<GRID(1255,1)>>>(H1, wq+woff[10], H2, 8, 1255, 1255, 2); // uw2 -> T32P
  ecg_qconv_direct<32,4><<<GRID(1255,4)>>>(H2, wq+woff[11], A1, 64, 1255, 1255, 0);// uw3 -> R32
  softmax(A1, A2, 1255, 0);                                                         // fp32 raw for ups
  ecg_qconv_ups<<<GRID(5016,1)>>>(A2, wq+woff[12], H1, 8, 5016, 4, 1255);          // uw4 -> T32P
  ecg_qconv_direct<32,1><<<GRID(5016,1)>>>(H1, wq+woff[13], A2, 4, 5016, 5016, 0); // uw5 -> R32
#undef GRID
  // ---- finalize ----
  {
    size_t total = (size_t)N * 5000;
    ecg_finalize<<<(unsigned)((total + 255) / 256), 256, 0, stream>>>(A2, (float*)d_out, N);
  }
}
